// TGAABlock_57939108823440
// MI455X (gfx1250) — matrix operand order repaired and checked by reading, not compiled, not run on hardware
//
#include <hip/hip_runtime.h>
#include <hip/hip_bf16.h>
#include <math.h>

typedef __attribute__((ext_vector_type(16))) _Float16 v16h;
typedef __attribute__((ext_vector_type(8)))  _Float16 v8h;
typedef __attribute__((ext_vector_type(8)))  float    v8f;

#define Bn    8
#define Cn    128
#define HEADS 2
#define HD    64
#define SRr   8
#define GRIDD 128
#define Nn    (GRIDD * GRIDD)    // 16384
#define NKV   256                // (128/8)^2
#define KCONV (Cn * SRr * SRr)   // 8192

static __device__ __forceinline__ v8f wmma16(v16h a, v16h b, v8f c) {
  // D = A(16x32 f16) * B(32x16 f16) + C(16x16 f32)
  return __builtin_amdgcn_wmma_f32_16x16x32_f16(false, a, false, b, (short)0, c,
                                                false, false);
}

// ---- operand packing per CDNA5 WMMA VGPR layouts ----
// A (16x32, f16): lane L holds row M=L&15; k-offset (L>>4)*8, elems j and j+16.
static __device__ __forceinline__ v16h pack_a_f16(const _Float16* src, int ld,
                                                  int kbase, int lane) {
  const int row = lane & 15, ko = (lane >> 4) * 8;
  const _Float16* p = src + (size_t)row * ld + kbase + ko;
  const v8h lo = *(const v8h*)(p);
  const v8h hi = *(const v8h*)(p + 16);
  v16h a;
#pragma unroll
  for (int j = 0; j < 8; ++j) { a[j] = lo[j]; a[j + 8] = hi[j]; }
  return a;
}
// A from f32 (LDS-resident tiles only)
static __device__ __forceinline__ v16h pack_a_f32(const float* src, int ld,
                                                  int kbase, int lane) {
  const int row = lane & 15, ko = (lane >> 4) * 8;
  const float* p = src + (size_t)row * ld + kbase + ko;
  v16h a;
#pragma unroll
  for (int j = 0; j < 8; ++j) {
    a[j]     = (_Float16)p[j];
    a[j + 8] = (_Float16)p[j + 16];
  }
  return a;
}
// B (32x16, f16): lane L holds col N=L&15; k-offset (L>>4)*8, elems j and j+16.
static __device__ __forceinline__ v16h pack_b_h16(const _Float16* base, int ld,
                                                  int coltile, int kbase,
                                                  int lane) {
  const int col = coltile * 16 + (lane & 15);
  const _Float16* p = base + (size_t)col * ld + kbase + (lane >> 4) * 8;
  const v8h lo = *(const v8h*)(p);
  const v8h hi = *(const v8h*)(p + 16);
  v16h b;
#pragma unroll
  for (int j = 0; j < 8; ++j) { b[j] = lo[j]; b[j + 8] = hi[j]; }
  return b;
}

static __device__ __forceinline__ float sigmoidf_(float x) {
  return 1.0f / (1.0f + expf(-x));
}

// ---------------------------------------------------------------------------
// Kernel 0: f32 -> f16 conversion (grid-strided, one-time staging)
// ---------------------------------------------------------------------------
__global__ void k_cvt(const float* __restrict__ src, _Float16* __restrict__ dst,
                      int n) {
  for (int i = blockIdx.x * blockDim.x + threadIdx.x; i < n;
       i += gridDim.x * blockDim.x)
    dst[i] = (_Float16)src[i];
}

// ---------------------------------------------------------------------------
// Kernel 1: dispersion gate  gate[b][c] = sigmoid( silu(gap@dg1)@dg2 )
// grid = B, block = 128 (one thread per channel)
// ---------------------------------------------------------------------------
__global__ void k_gate(const float* __restrict__ x,
                       const float* __restrict__ dg1_w, const float* __restrict__ dg1_b,
                       const float* __restrict__ dg2_w, const float* __restrict__ dg2_b,
                       float* __restrict__ gate) {
  __shared__ float gap[Cn];
  __shared__ float g1[Cn];
  const int b = blockIdx.x, c = threadIdx.x;
  const float* xb = x + (size_t)b * Nn * Cn;
  float s = 0.0f;
  for (int n = 0; n < Nn; ++n) s += xb[(size_t)n * Cn + c];
  gap[c] = s * (1.0f / (float)Nn);
  __syncthreads();
  float a = dg1_b[c];
  for (int k = 0; k < Cn; ++k) a += gap[k] * dg1_w[c * Cn + k];
  g1[c] = a * sigmoidf_(a);  // silu
  __syncthreads();
  float o = dg2_b[c];
  for (int k = 0; k < Cn; ++k) o += g1[k] * dg2_w[c * Cn + k];
  gate[b * Cn + c] = sigmoidf_(o);
}

// ---------------------------------------------------------------------------
// Kernel 2: intensity gate  iw[b][n][h]  (pre-softmax logit scaling)
// grid = B*N/256, block = 256
// ---------------------------------------------------------------------------
__global__ void k_iw(const float* __restrict__ ti,
                     const float* __restrict__ ip1_w, const float* __restrict__ ip1_b,
                     const float* __restrict__ ip2_w, const float* __restrict__ ip2_b,
                     float* __restrict__ iw) {
  const int idx = blockIdx.x * blockDim.x + threadIdx.x;  // token over B*N
  const float t = ti[idx];
  float a0 = ip2_b[0], a1 = ip2_b[1];
  for (int c = 0; c < Cn; ++c) {
    float u = t * ip1_w[c] + ip1_b[c];
    u = u * sigmoidf_(u);  // silu
    a0 += u * ip2_w[0 * Cn + c];
    a1 += u * ip2_w[1 * Cn + c];
  }
  iw[idx * 2 + 0] = sigmoidf_(a0);
  iw[idx * 2 + 1] = sigmoidf_(a1);
}

// ---------------------------------------------------------------------------
// Kernel 3: Q projection (WMMA, all-f16 operands)
// grid = (B*N/16)/8, block = 256 (8 waves, one 16-row tile each)
// ---------------------------------------------------------------------------
__global__ void k_qproj(const _Float16* __restrict__ xh,
                        const _Float16* __restrict__ q_wh,
                        const float* __restrict__ q_b,
                        _Float16* __restrict__ q_ws) {
  const int lane = threadIdx.x & 31;
  const int wid  = blockIdx.x * 8 + (threadIdx.x >> 5);
  const size_t row0 = (size_t)wid * 16;
  const _Float16* arow = xh + row0 * Cn;

  v16h a[4];
#pragma unroll
  for (int kc = 0; kc < 4; ++kc) a[kc] = pack_a_f16(arow, Cn, kc * 32, lane);

#pragma unroll
  for (int ct = 0; ct < 8; ++ct) {
    const int col = ct * 16 + (lane & 15);
    v8f acc;
#pragma unroll
    for (int r = 0; r < 8; ++r) acc[r] = q_b[col];
#pragma unroll
    for (int kc = 0; kc < 4; ++kc)
      acc = wmma16(a[kc], pack_b_h16(q_wh, Cn, ct, kc * 32, lane), acc);
#pragma unroll
    for (int r = 0; r < 8; ++r) {
      const int row = r + 8 * (lane >> 4);
      q_ws[(row0 + row) * Cn + col] = (_Float16)acc[r];
    }
  }
}

// ---------------------------------------------------------------------------
// Kernel 4: SR conv (im2col WMMA, K=8192) + bias + LayerNorm + KV proj (WMMA)
// writes K[b][h][m][d] (row-major) and Vt[b][h][d][m] (transposed) as f16.
// grid = B*16 (one wave per 16 kv tokens), block = 32
// ---------------------------------------------------------------------------
__global__ void k_kv(const _Float16* __restrict__ xh,
                     const _Float16* __restrict__ sr_wh,
                     const float* __restrict__ sr_b,
                     const float* __restrict__ ln_g, const float* __restrict__ ln_b,
                     const _Float16* __restrict__ kv_wh,
                     const float* __restrict__ kv_b,
                     _Float16* __restrict__ Kws, _Float16* __restrict__ Vtws) {
  __shared__ float xsr[16][Cn];  // 8 KB
  const int lane = threadIdx.x & 31;
  const int b = blockIdx.x >> 4, tile = blockIdx.x & 15;
  const int row = lane & 15, ko = (lane >> 4) * 8;
  const int pos = tile * 16 + row, py = pos >> 4, px = pos & 15;
  const _Float16* xb = xh + (size_t)b * Nn * Cn;

  v8f acc[8];
#pragma unroll
  for (int ct = 0; ct < 8; ++ct) {
    const int col = ct * 16 + (lane & 15);
#pragma unroll
    for (int r = 0; r < 8; ++r) acc[ct][r] = sr_b[col];
  }

  for (int chunk = 0; chunk < KCONV / 32; ++chunk) {
    const int kbase = chunk * 32;
    v16h a;
#pragma unroll
    for (int j = 0; j < 8; ++j) {
#pragma unroll
      for (int half = 0; half < 2; ++half) {
        const int k  = kbase + ko + j + half * 16;
        const int ci = k >> 6, rem = k & 63, ky = rem >> 3, kx = rem & 7;
        const int n  = (py * SRr + ky) * GRIDD + px * SRr + kx;
        a[j + half * 8] = xb[(size_t)n * Cn + ci];
      }
    }
#pragma unroll
    for (int ct = 0; ct < 8; ++ct)
      acc[ct] = wmma16(a, pack_b_h16(sr_wh, KCONV, ct, kbase, lane), acc[ct]);
  }

  // spill conv result to LDS
#pragma unroll
  for (int ct = 0; ct < 8; ++ct) {
    const int col = ct * 16 + (lane & 15);
#pragma unroll
    for (int r = 0; r < 8; ++r) xsr[r + 8 * (lane >> 4)][col] = acc[ct][r];
  }
  __syncthreads();

  // LayerNorm per kv token (lanes 0..15, one row each)
  if (lane < 16) {
    float mu = 0.0f;
    for (int c = 0; c < Cn; ++c) mu += xsr[lane][c];
    mu *= (1.0f / Cn);
    float var = 0.0f;
    for (int c = 0; c < Cn; ++c) { float d = xsr[lane][c] - mu; var += d * d; }
    var *= (1.0f / Cn);
    const float rs = rsqrtf(var + 1e-5f);
    for (int c = 0; c < Cn; ++c)
      xsr[lane][c] = (xsr[lane][c] - mu) * rs * ln_g[c] + ln_b[c];
  }
  __syncthreads();

  // KV projection: (16 x 128) @ kv_w.T -> 16 x 256
  v16h a2[4];
#pragma unroll
  for (int kc = 0; kc < 4; ++kc) a2[kc] = pack_a_f32(&xsr[0][0], Cn, kc * 32, lane);

#pragma unroll
  for (int ct = 0; ct < 16; ++ct) {
    const int col = ct * 16 + (lane & 15);
    v8f o;
#pragma unroll
    for (int r = 0; r < 8; ++r) o[r] = kv_b[col];
#pragma unroll
    for (int kc = 0; kc < 4; ++kc)
      o = wmma16(a2[kc], pack_b_h16(kv_wh, Cn, ct, kc * 32, lane), o);
#pragma unroll
    for (int r = 0; r < 8; ++r) {
      const int m = tile * 16 + r + 8 * (lane >> 4);
      if (col < Cn) {  // K half
        const int h = col / HD, d = col % HD;
        Kws[(((size_t)b * HEADS + h) * NKV + m) * HD + d] = (_Float16)o[r];
      } else {         // V half, stored transposed [d][m]
        const int jj = col - Cn, h = jj / HD, d = jj % HD;
        Vtws[(((size_t)b * HEADS + h) * HD + d) * NKV + m] = (_Float16)o[r];
      }
    }
  }
}

// ---------------------------------------------------------------------------
// Kernel 5: fused attention. One wave per 16 query rows.
// S = QK^T * scale * iw  -> softmax -> P V -> * dispersion gate -> f16 out
// grid = B*HEADS*(N/64), block = 128 (4 waves)
// ---------------------------------------------------------------------------
__global__ void k_attn(const _Float16* __restrict__ q_ws,
                       const _Float16* __restrict__ Kws,
                       const _Float16* __restrict__ Vtws,
                       const float* __restrict__ iw,
                       const float* __restrict__ gate,
                       _Float16* __restrict__ attn_out) {
  __shared__ float S[4][16][NKV];  // 64 KB
  const int lane = threadIdx.x & 31, w = threadIdx.x >> 5;
  const int bh = blockIdx.x / (Nn / 64);
  const int b = bh / HEADS, h = bh % HEADS;
  const int n0 = (blockIdx.x % (Nn / 64)) * 64 + w * 16;
  const float scale = rsqrtf((float)HD);

  // Q tile: rows n0..n0+15, cols h*64..h*64+63 -> two 16x32 A operands
  const _Float16* qbase = q_ws + ((size_t)b * Nn + n0) * Cn + h * HD;
  v16h aq[2];
#pragma unroll
  for (int kc = 0; kc < 2; ++kc) aq[kc] = pack_a_f16(qbase, Cn, kc * 32, lane);

  const _Float16* Kb = Kws + ((size_t)b * HEADS + h) * NKV * HD;
#pragma unroll
  for (int ct = 0; ct < 16; ++ct) {
    v8f s = {};
#pragma unroll
    for (int kc = 0; kc < 2; ++kc)
      s = wmma16(aq[kc], pack_b_h16(Kb, HD, ct, kc * 32, lane), s);
    const int col = ct * 16 + (lane & 15);
#pragma unroll
    for (int r = 0; r < 8; ++r) S[w][r + 8 * (lane >> 4)][col] = s[r];
  }

  // per-row softmax with combined scale: lanes 0..15 own one row each
  if (lane < 16) {
    const float g = scale * iw[((size_t)b * Nn + n0 + lane) * HEADS + h];
    float mx = -3.4e38f;
    for (int j = 0; j < NKV; ++j) {
      const float v = S[w][lane][j] * g;
      S[w][lane][j] = v;
      mx = fmaxf(mx, v);
    }
    float sum = 0.0f;
    for (int j = 0; j < NKV; ++j) {
      const float e = expf(S[w][lane][j] - mx);
      S[w][lane][j] = e;
      sum += e;
    }
    const float inv = 1.0f / sum;
    for (int j = 0; j < NKV; ++j) S[w][lane][j] *= inv;
  }
  __builtin_amdgcn_wave_barrier();

  // O = P(16x256) @ V(256x64), V read transposed (contiguous in m)
  const _Float16* Vb = Vtws + ((size_t)b * HEADS + h) * HD * NKV;
#pragma unroll
  for (int nt = 0; nt < 4; ++nt) {
    v8f o = {};
#pragma unroll
    for (int kc = 0; kc < 8; ++kc)
      o = wmma16(pack_a_f32(&S[w][0][0], NKV, kc * 32, lane),
                 pack_b_h16(Vb, NKV, nt, kc * 32, lane), o);
    const int ch = h * HD + nt * 16 + (lane & 15);
    const float og = gate[b * Cn + ch];
#pragma unroll
    for (int r = 0; r < 8; ++r) {
      const int row = r + 8 * (lane >> 4);
      attn_out[((size_t)b * Nn + n0 + row) * Cn + ch] = (_Float16)(o[r] * og);
    }
  }
}

// ---------------------------------------------------------------------------
// Kernel 6: output projection (WMMA, all-f16 operands) -> f32 d_out
// grid = (B*N/16)/8, block = 256
// ---------------------------------------------------------------------------
__global__ void k_proj(const _Float16* __restrict__ attn_out,
                       const _Float16* __restrict__ proj_wh,
                       const float* __restrict__ proj_b,
                       float* __restrict__ out) {
  const int lane = threadIdx.x & 31;
  const int wid  = blockIdx.x * 8 + (threadIdx.x >> 5);
  const size_t row0 = (size_t)wid * 16;
  const _Float16* arow = attn_out + row0 * Cn;

  v16h a[4];
#pragma unroll
  for (int kc = 0; kc < 4; ++kc) a[kc] = pack_a_f16(arow, Cn, kc * 32, lane);

#pragma unroll
  for (int ct = 0; ct < 8; ++ct) {
    const int col = ct * 16 + (lane & 15);
    v8f acc;
#pragma unroll
    for (int r = 0; r < 8; ++r) acc[r] = proj_b[col];
#pragma unroll
    for (int kc = 0; kc < 4; ++kc)
      acc = wmma16(a[kc], pack_b_h16(proj_wh, Cn, ct, kc * 32, lane), acc);
#pragma unroll
    for (int r = 0; r < 8; ++r) {
      const int row = r + 8 * (lane >> 4);
      out[(row0 + row) * Cn + col] = acc[r];
    }
  }
}

// ---------------------------------------------------------------------------
extern "C" void kernel_launch(void* const* d_in, const int* in_sizes, int n_in,
                              void* d_out, int out_size, void* d_ws, size_t ws_size,
                              hipStream_t stream) {
  (void)in_sizes; (void)n_in; (void)out_size; (void)ws_size;
  const float* x      = (const float*)d_in[0];
  const float* ti     = (const float*)d_in[1];
  const float* q_w    = (const float*)d_in[2];
  const float* q_b    = (const float*)d_in[3];
  const float* kv_w   = (const float*)d_in[4];
  const float* kv_b   = (const float*)d_in[5];
  const float* sr_w   = (const float*)d_in[6];
  const float* sr_b   = (const float*)d_in[7];
  const float* ln_g   = (const float*)d_in[8];
  const float* ln_b   = (const float*)d_in[9];
  const float* ip1_w  = (const float*)d_in[10];
  const float* ip1_b  = (const float*)d_in[11];
  const float* ip2_w  = (const float*)d_in[12];
  const float* ip2_b  = (const float*)d_in[13];
  const float* dg1_w  = (const float*)d_in[14];
  const float* dg1_b  = (const float*)d_in[15];
  const float* dg2_w  = (const float*)d_in[16];
  const float* dg2_b  = (const float*)d_in[17];
  const float* proj_w = (const float*)d_in[18];
  const float* proj_b = (const float*)d_in[19];
  float* outp = (float*)d_out;

  // workspace layout
  char* ws = (char*)d_ws;
  size_t off = 0;
  _Float16* q_ws     = (_Float16*)(ws + off); off += (size_t)Bn * Nn * Cn * 2;  // 32 MB
  _Float16* attn_ws  = (_Float16*)(ws + off); off += (size_t)Bn * Nn * Cn * 2;  // 32 MB
  _Float16* xh       = (_Float16*)(ws + off); off += (size_t)Bn * Nn * Cn * 2;  // 32 MB
  _Float16* sr_wh    = (_Float16*)(ws + off); off += (size_t)Cn * KCONV * 2;    //  2 MB
  _Float16* q_wh     = (_Float16*)(ws + off); off += (size_t)Cn * Cn * 2;
  _Float16* kv_wh    = (_Float16*)(ws + off); off += (size_t)2 * Cn * Cn * 2;
  _Float16* proj_wh  = (_Float16*)(ws + off); off += (size_t)Cn * Cn * 2;
  _Float16* Kws      = (_Float16*)(ws + off); off += (size_t)Bn * HEADS * NKV * HD * 2;
  _Float16* Vtws     = (_Float16*)(ws + off); off += (size_t)Bn * HEADS * HD * NKV * 2;
  float*    iw_ws    = (float*)(ws + off);    off += (size_t)Bn * Nn * HEADS * 4;  // 1 MB
  float*    gate_ws  = (float*)(ws + off);    off += (size_t)Bn * Cn * 4;

  // one-time f32 -> f16 staging (x + all GEMM weights)
  k_cvt<<<2048, 256, 0, stream>>>(x, xh, Bn * Nn * Cn);
  k_cvt<<<256, 256, 0, stream>>>(sr_w, sr_wh, Cn * KCONV);
  k_cvt<<<16, 256, 0, stream>>>(q_w, q_wh, Cn * Cn);
  k_cvt<<<32, 256, 0, stream>>>(kv_w, kv_wh, 2 * Cn * Cn);
  k_cvt<<<16, 256, 0, stream>>>(proj_w, proj_wh, Cn * Cn);

  k_gate <<<Bn, Cn, 0, stream>>>(x, dg1_w, dg1_b, dg2_w, dg2_b, gate_ws);
  k_iw   <<<(Bn * Nn) / 256, 256, 0, stream>>>(ti, ip1_w, ip1_b, ip2_w, ip2_b, iw_ws);
  k_qproj<<<(Bn * Nn / 16) / 8, 256, 0, stream>>>(xh, q_wh, q_b, q_ws);
  k_kv   <<<Bn * 16, 32, 0, stream>>>(xh, sr_wh, sr_b, ln_g, ln_b, kv_wh, kv_b, Kws, Vtws);
  k_attn <<<Bn * HEADS * (Nn / 64), 128, 0, stream>>>(q_ws, Kws, Vtws, iw_ws, gate_ws, attn_ws);
  k_proj <<<(Bn * Nn / 16) / 8, 256, 0, stream>>>(attn_ws, proj_wh, proj_b, outp);
}
